// NoiseBlockMoE_27195732918431
// MI455X (gfx1250) — matrix operand order repaired and checked by reading, not compiled, not run on hardware
//
#include <hip/hip_runtime.h>
#include <hip/hip_bf16.h>
#include <math.h>

typedef __attribute__((ext_vector_type(16))) __bf16 v16bf;
typedef __attribute__((ext_vector_type(8)))  float  v8f;

namespace cfg {
constexpr int B = 4, T = 1024, D = 1024, H = 16, DH = 64, E = 8, FF = 4096;
constexpr int BT = B * T;
}

struct U128 { unsigned int x, y, z, w; };
union AFrag { v16bf v; __bf16 e[16]; U128 q[2]; };

// ---- CDNA5 async global->LDS copy (ASYNCcnt-tracked) -----------------------
// lds_off: workgroup-relative LDS byte offset (low 32 bits of generic pointer
// to a __shared__ object on amdgcn); gaddr: 64-bit global address.
static __device__ inline void async_copy_b128(unsigned lds_off, const void* gaddr) {
  asm volatile("global_load_async_to_lds_b128 %0, %1, off"
               :
               : "v"(lds_off), "v"(gaddr)
               : "memory");
}
static __device__ inline void wait_async_le2() {
  asm volatile("s_wait_asynccnt 0x2" ::: "memory");
}
static __device__ inline void wait_async_le0() {
  asm volatile("s_wait_asynccnt 0x0" ::: "memory");
}

// ---- WMMA fragment loaders (bf16, 16x16x32) -------------------------------
// A: 16(M) x 32(K), row-major [M, lda]. Lane L<16 holds row M=L with
// K = {k0..k0+7} U {k0+16..k0+23}; lane L>=16 holds same row, K base +8.
static __device__ inline v16bf load_a_frag(const __bf16* __restrict__ A, int lda,
                                           int m0, int k0, int lane) {
  AFrag f;
  const int m  = m0 + (lane & 15);
  const int kb = k0 + ((lane >> 4) << 3);
  const __bf16* row = A + (size_t)m * lda;
  f.q[0] = *(const U128*)(row + kb);
  f.q[1] = *(const U128*)(row + kb + 16);
  return f.v;
}

// B: 32(K) x 16(N), stored as Bt row-major [N, ldb] (i.e. weight W[n][k]).
// Lane L<16 holds column N=L with K = {k0..k0+7} U {k0+16..k0+23}; lane L>=16 holds K base +8.
// Works for both global and LDS source pointers (LDS -> ds_load_b128).
static __device__ inline v16bf load_b_frag(const __bf16* __restrict__ Bt, int ldb,
                                           int n0, int k0, int lane) {
  AFrag f;
  const int n  = n0 + (lane & 15);
  const int kb = k0 + ((lane >> 4) << 3);
  const __bf16* row = Bt + (size_t)n * ldb;
  f.q[0] = *(const U128*)(row + kb);
  f.q[1] = *(const U128*)(row + kb + 16);
  return f.v;
}

static __device__ inline v8f wmma_bf16(v16bf a, v16bf b, v8f c) {
  return __builtin_amdgcn_wmma_f32_16x16x32_bf16(false, a, false, b, (short)0, c,
                                                 false, false);
}

// ---- Block-cooperative GEMM tile with double-buffered async B staging ------
// Block: 256 threads = 8 waves stacked in M (16 rows each), all sharing one
// 64-column N tile. Per 64-wide K step the block stages a 64x64 bf16 B tile
// (8 KB) into LDS: each thread issues two async B128s (one wave = 2 async
// instructions per tile). Double-buffered: issue tile k+1, wait asynccnt<=2
// (in-order => tile k landed), barrier, compute 2 K-substeps (8 WMMAs),
// barrier. Barrier/wait cost amortized over 8 WMMAs per wave.
static __device__ inline void gemm_tile_async(
    const __bf16* __restrict__ A, int lda,
    const __bf16* __restrict__ Bt, int ldb, int K,
    int m0, int n0blk, int lane, int tid,
    __bf16* Bs /* [2][64][64] in LDS */, v8f acc[4]) {
  const int trow = tid >> 2;  // 0..63: N row within block tile
  const int kseg = tid & 3;   // 0..3 : 8-element K segment (and +32)
  const __bf16* gsrc = Bt + (size_t)(n0blk + trow) * ldb + kseg * 8;
  const unsigned lds_base = (unsigned)(uintptr_t)Bs;
  const unsigned row_off = (unsigned)(trow * 64 + kseg * 8) * 2u;
  const unsigned buf_stride = 64u * 64u * 2u;  // 8 KB

  // prologue: stage tile 0 into buffer 0
  async_copy_b128(lds_base + row_off, gsrc);
  async_copy_b128(lds_base + row_off + 64u, gsrc + 32);

  const int NK = K / 64;
  for (int kt = 0; kt < NK; ++kt) {
    const int buf = kt & 1;
    if (kt + 1 < NK) {
      const unsigned dst = lds_base + (unsigned)(buf ^ 1) * buf_stride + row_off;
      const __bf16* src = gsrc + (size_t)(kt + 1) * 64;
      async_copy_b128(dst, src);
      async_copy_b128(dst + 64u, src + 32);
      wait_async_le2();
    } else {
      wait_async_le0();
    }
    __syncthreads();
    const __bf16* Bl = Bs + (size_t)buf * (64 * 64);
    if (kt + 2 < NK)
      __builtin_prefetch(A + (size_t)(m0 + (lane & 15)) * lda + (kt + 2) * 64);
#pragma unroll
    for (int ks = 0; ks < 64; ks += 32) {
      v16bf a = load_a_frag(A, lda, m0, kt * 64 + ks, lane);
#pragma unroll
      for (int j = 0; j < 4; ++j) {
        v16bf b = load_b_frag(Bl, 64, j * 16, ks, lane);
        acc[j] = wmma_bf16(a, b, acc[j]);
      }
    }
    __syncthreads();  // protect buffer (buf) before it is overwritten
  }
}

// ---- f32 -> bf16 conversion ------------------------------------------------
__global__ __launch_bounds__(256) void k_cvt(const float* __restrict__ s,
                                             __bf16* __restrict__ d, size_t n) {
  size_t i = (size_t)blockIdx.x * 256 + threadIdx.x;
  const size_t stride = (size_t)gridDim.x * 256;
  for (; i < n; i += stride) d[i] = (__bf16)s[i];
}

// ---- LayerNorm (optionally + broadcast c), outputs bf16 and/or f32 ---------
__global__ __launch_bounds__(256) void k_ln(const float* __restrict__ x,
                                            const float* __restrict__ cadd,
                                            const float* __restrict__ w,
                                            __bf16* __restrict__ outb,
                                            float* __restrict__ outf) {
  using namespace cfg;
  __shared__ float red[256];
  const int row = blockIdx.x;
  const int tid = threadIdx.x;
  const float* xr = x + (size_t)row * D;
  float vals[4];
  float s = 0.f;
#pragma unroll
  for (int i = 0; i < 4; ++i) { vals[i] = xr[tid + 256 * i]; s += vals[i]; }
  red[tid] = s;
  __syncthreads();
  for (int off = 128; off > 0; off >>= 1) {
    if (tid < off) red[tid] += red[tid + off];
    __syncthreads();
  }
  const float mu = red[0] * (1.f / D);
  __syncthreads();
  float s2 = 0.f;
#pragma unroll
  for (int i = 0; i < 4; ++i) { const float d0 = vals[i] - mu; s2 += d0 * d0; }
  red[tid] = s2;
  __syncthreads();
  for (int off = 128; off > 0; off >>= 1) {
    if (tid < off) red[tid] += red[tid + off];
    __syncthreads();
  }
  const float rs = rsqrtf(red[0] * (1.f / D) + 1e-5f);
  const float* cr = cadd ? cadd + (size_t)(row / T) * D : nullptr;
#pragma unroll
  for (int i = 0; i < 4; ++i) {
    const int idx = tid + 256 * i;
    float y = (vals[i] - mu) * rs * w[idx];
    if (cr) y += cr[idx];
    if (outb) outb[(size_t)row * D + idx] = (__bf16)y;
    if (outf) outf[(size_t)row * D + idx] = y;
  }
}

// ---- QKV GEMM: [BT,D] x Wqkv^T -> scatter into q/k/v [B,H,T,DH] ------------
__global__ __launch_bounds__(256) void k_gemm_qkv(
    const __bf16* __restrict__ hB, const __bf16* __restrict__ Wq,
    __bf16* __restrict__ q, __bf16* __restrict__ k, __bf16* __restrict__ v) {
  using namespace cfg;
  __shared__ __bf16 Bs[2][64][64];
  const int lane = threadIdx.x & 31, wave = threadIdx.x >> 5;
  const int m0 = blockIdx.y * 128 + wave * 16;
  const int n0 = blockIdx.x * 64;
  v8f zero = {};
  v8f acc[4] = {zero, zero, zero, zero};
  gemm_tile_async(hB, D, Wq, D, D, m0, n0, lane, threadIdx.x, &Bs[0][0][0], acc);
  const int half = lane >> 4, l15 = lane & 15;
#pragma unroll
  for (int j = 0; j < 4; ++j) {
#pragma unroll
    for (int vi = 0; vi < 8; ++vi) {
      const int row = m0 + vi + half * 8;
      const int col = n0 + j * 16 + l15;
      const int which = col >> 10;
      const int d = col & (D - 1);
      const int hh = d >> 6, dh = d & 63;
      const int t = row & (T - 1), b = row >> 10;
      __bf16* dst = (which == 0) ? q : (which == 1) ? k : v;
      dst[(((size_t)b * H + hh) * T + t) * DH + dh] = (__bf16)acc[j][vi];
    }
  }
}

// ---- Flash attention: one wave per (b,h, 16-query tile) --------------------
__global__ __launch_bounds__(32) void k_attn(const __bf16* __restrict__ Q,
                                             const __bf16* __restrict__ Kb,
                                             const __bf16* __restrict__ Vb,
                                             __bf16* __restrict__ O) {
  using namespace cfg;
  __shared__ __bf16 pls[16][40];   // P tile staging for D-frag -> A-frag transpose
  const int lane = threadIdx.x & 31;
  const int half = lane >> 4;
  const int l15 = lane & 15;
  const int qt = blockIdx.x;
  const int bh = blockIdx.y;
  const int b = bh / H, h = bh % H;
  const __bf16* qp = Q + (size_t)bh * T * DH;
  const __bf16* kp = Kb + (size_t)bh * T * DH;
  const __bf16* vp = Vb + (size_t)bh * T * DH;

  const v16bf qa0 = load_a_frag(qp, DH, qt * 16, 0, lane);
  const v16bf qa1 = load_a_frag(qp, DH, qt * 16, 32, lane);

  v8f zero = {};
  v8f oacc[4] = {zero, zero, zero, zero};
  float m_r[8], l_r[8];
#pragma unroll
  for (int i = 0; i < 8; ++i) { m_r[i] = -1e30f; l_r[i] = 0.f; }

  const int kt_max = (qt * 16 + 15) >> 5;
  for (int kt = 0; kt <= kt_max; ++kt) {
    const int n0 = kt * 32;
    v8f s0 = zero, s1 = zero;
    {
      v16bf b0a = load_b_frag(kp, DH, n0, 0, lane);
      v16bf b0b = load_b_frag(kp, DH, n0, 32, lane);
      s0 = wmma_bf16(qa0, b0a, s0);
      s0 = wmma_bf16(qa1, b0b, s0);
      v16bf b1a = load_b_frag(kp, DH, n0 + 16, 0, lane);
      v16bf b1b = load_b_frag(kp, DH, n0 + 16, 32, lane);
      s1 = wmma_bf16(qa0, b1a, s1);
      s1 = wmma_bf16(qa1, b1b, s1);
    }
    const float scale = 0.125f;   // 1/sqrt(64)
    float alpha[8];
#pragma unroll
    for (int vi = 0; vi < 8; ++vi) {
      const int qg = qt * 16 + vi + half * 8;
      const int c0 = n0 + l15;
      const int c1 = n0 + 16 + l15;
      float a0 = (c0 <= qg) ? s0[vi] * scale : -1e30f;
      float a1 = (c1 <= qg) ? s1[vi] * scale : -1e30f;
      float mx = fmaxf(a0, a1);
#pragma unroll
      for (int off = 1; off < 16; off <<= 1)
        mx = fmaxf(mx, __shfl_xor(mx, off, 32));
      const float mnew = fmaxf(m_r[vi], mx);
      alpha[vi] = expf(m_r[vi] - mnew);
      const float p0 = (a0 > -1e29f) ? expf(a0 - mnew) : 0.f;
      const float p1 = (a1 > -1e29f) ? expf(a1 - mnew) : 0.f;
      float rsum = p0 + p1;
#pragma unroll
      for (int off = 1; off < 16; off <<= 1)
        rsum += __shfl_xor(rsum, off, 32);
      l_r[vi] = l_r[vi] * alpha[vi] + rsum;
      m_r[vi] = mnew;
      pls[vi + half * 8][l15] = (__bf16)p0;
      pls[vi + half * 8][16 + l15] = (__bf16)p1;
    }
#pragma unroll
    for (int j = 0; j < 4; ++j)
#pragma unroll
      for (int vi = 0; vi < 8; ++vi) oacc[j][vi] *= alpha[vi];
    __syncthreads();
    AFrag pa;
    {
      const int kb = half * 8;
#pragma unroll
      for (int j = 0; j < 8; ++j) {
        pa.e[j]     = pls[l15][kb + j];
        pa.e[j + 8] = pls[l15][kb + 16 + j];
      }
    }
    __syncthreads();
#pragma unroll
    for (int j = 0; j < 4; ++j) {
      AFrag vbf;
      const int dh = j * 16 + l15;
      const int kbase = half * 8;
#pragma unroll
      for (int jj = 0; jj < 16; ++jj)
        vbf.e[jj] = vp[(size_t)(n0 + kbase + (jj & 7) + ((jj >> 3) << 4)) * DH + dh];
      oacc[j] = wmma_bf16(pa.v, vbf.v, oacc[j]);
    }
  }
#pragma unroll
  for (int vi = 0; vi < 8; ++vi) {
    const int t = qt * 16 + vi + half * 8;
    const float inv_l = 1.f / l_r[vi];
    __bf16* orow = O + ((size_t)b * T + t) * D + h * DH;
#pragma unroll
    for (int j = 0; j < 4; ++j)
      orow[j * 16 + l15] = (__bf16)(oacc[j][vi] * inv_l);
  }
}

// ---- Proj GEMM + residual: x1 = x + o @ Wp^T -------------------------------
__global__ __launch_bounds__(256) void k_proj(const __bf16* __restrict__ o,
                                              const __bf16* __restrict__ Wp,
                                              const float* __restrict__ x,
                                              float* __restrict__ x1) {
  using namespace cfg;
  __shared__ __bf16 Bs[2][64][64];
  const int lane = threadIdx.x & 31, wave = threadIdx.x >> 5;
  const int m0 = blockIdx.y * 128 + wave * 16;
  const int n0 = blockIdx.x * 64;
  v8f zero = {};
  v8f acc[4] = {zero, zero, zero, zero};
  gemm_tile_async(o, D, Wp, D, D, m0, n0, lane, threadIdx.x, &Bs[0][0][0], acc);
  const int half = lane >> 4, l15 = lane & 15;
#pragma unroll
  for (int j = 0; j < 4; ++j)
#pragma unroll
    for (int vi = 0; vi < 8; ++vi) {
      const size_t row = m0 + vi + half * 8;
      const int col = n0 + j * 16 + l15;
      x1[row * D + col] = x[row * D + col] + acc[j][vi];
    }
}

// ---- Router: 8 logits/token, softmax, eps-clamp, top-2 renorm --------------
__global__ __launch_bounds__(256) void k_router(const float* __restrict__ x2,
                                                const float* __restrict__ rw,
                                                float* __restrict__ probs) {
  using namespace cfg;
  __shared__ float lg[E];
  const int row = blockIdx.x;
  const int e = threadIdx.x >> 5;
  const int lane = threadIdx.x & 31;
  const float* xr = x2 + (size_t)row * D;
  const float* wr = rw + (size_t)e * D;
  float s = 0.f;
  for (int i = lane; i < D; i += 32) s += xr[i] * wr[i];
#pragma unroll
  for (int off = 16; off > 0; off >>= 1) s += __shfl_xor(s, off, 32);
  if (lane == 0) lg[e] = s;
  __syncthreads();
  if (threadIdx.x == 0) {
    float mx = lg[0];
    for (int i = 1; i < E; ++i) mx = fmaxf(mx, lg[i]);
    float p[E]; float sum = 0.f;
    for (int i = 0; i < E; ++i) { p[i] = expf(lg[i] - mx); sum += p[i]; }
    const float inv = 1.f / sum;
    for (int i = 0; i < E; ++i) {
      p[i] = p[i] * inv + 1e-9f;
      p[i] = fminf(fmaxf(p[i], 1e-9f), 1.f - 1e-9f);
    }
    int i1 = 0;
    for (int i = 1; i < E; ++i) if (p[i] > p[i1]) i1 = i;
    int i2 = (i1 == 0) ? 1 : 0;
    for (int i = 0; i < E; ++i) if (i != i1 && p[i] > p[i2]) i2 = i;
    const float denom = 1.f / (p[i1] + p[i2]);
    for (int i = 0; i < E; ++i)
      probs[(size_t)row * E + i] = (i == i1 || i == i2) ? p[i] * denom : 0.f;
  }
}

// ---- Expert FC1: hid = gelu_exact(x2 @ w1[e]^T) ----------------------------
__global__ __launch_bounds__(256) void k_fc1(const __bf16* __restrict__ xb,
                                             const __bf16* __restrict__ W1,
                                             __bf16* __restrict__ hid) {
  using namespace cfg;
  __shared__ __bf16 Bs[2][64][64];
  const int lane = threadIdx.x & 31, wave = threadIdx.x >> 5;
  const int m0 = blockIdx.y * 128 + wave * 16;
  const int n0 = blockIdx.x * 64;
  v8f zero = {};
  v8f acc[4] = {zero, zero, zero, zero};
  gemm_tile_async(xb, D, W1, D, D, m0, n0, lane, threadIdx.x, &Bs[0][0][0], acc);
  const int half = lane >> 4, l15 = lane & 15;
#pragma unroll
  for (int j = 0; j < 4; ++j)
#pragma unroll
    for (int vi = 0; vi < 8; ++vi) {
      const size_t row = m0 + vi + half * 8;
      const int col = n0 + j * 16 + l15;
      const float u = acc[j][vi];
      const float g = 0.5f * u * (1.f + erff(u * 0.70710678118654752f));
      hid[row * FF + col] = (__bf16)g;
    }
}

// ---- Expert FC2: out += prob[:,e] * (hid @ w2[e]^T) ------------------------
__global__ __launch_bounds__(256) void k_fc2(const __bf16* __restrict__ hid,
                                             const __bf16* __restrict__ W2,
                                             const float* __restrict__ probs,
                                             int e, float* __restrict__ out) {
  using namespace cfg;
  __shared__ __bf16 Bs[2][64][64];
  const int lane = threadIdx.x & 31, wave = threadIdx.x >> 5;
  const int m0 = blockIdx.y * 128 + wave * 16;
  const int n0 = blockIdx.x * 64;
  v8f zero = {};
  v8f acc[4] = {zero, zero, zero, zero};
  gemm_tile_async(hid, FF, W2, FF, FF, m0, n0, lane, threadIdx.x, &Bs[0][0][0], acc);
  const int half = lane >> 4, l15 = lane & 15;
#pragma unroll
  for (int j = 0; j < 4; ++j)
#pragma unroll
    for (int vi = 0; vi < 8; ++vi) {
      const size_t row = m0 + vi + half * 8;
      const int col = n0 + j * 16 + l15;
      const float pr = probs[row * E + e];
      out[row * D + col] += pr * acc[j][vi];
    }
}

extern "C" void kernel_launch(void* const* d_in, const int* in_sizes, int n_in,
                              void* d_out, int out_size, void* d_ws, size_t ws_size,
                              hipStream_t stream) {
  using namespace cfg;
  (void)in_sizes; (void)n_in; (void)out_size; (void)ws_size;
  const float* x        = (const float*)d_in[0];
  const float* c        = (const float*)d_in[1];
  const float* ln1_w    = (const float*)d_in[2];
  const float* w_qkv    = (const float*)d_in[3];
  const float* w_proj   = (const float*)d_in[4];
  const float* ln2_w    = (const float*)d_in[5];
  const float* router_w = (const float*)d_in[6];
  const float* e_w1     = (const float*)d_in[7];
  const float* e_w2     = (const float*)d_in[8];
  float* out = (float*)d_out;

  char* ws = (char*)d_ws;
  size_t off = 0;
  auto alloc = [&](size_t bytes) -> char* {
    char* p = ws + off;
    off += (bytes + 255) & ~(size_t)255;
    return p;
  };
  __bf16* hb     = (__bf16*)alloc((size_t)BT * D * 2);
  __bf16* qb     = (__bf16*)alloc((size_t)BT * D * 2);
  __bf16* kb     = (__bf16*)alloc((size_t)BT * D * 2);
  __bf16* vb     = (__bf16*)alloc((size_t)BT * D * 2);
  __bf16* ob     = (__bf16*)alloc((size_t)BT * D * 2);
  float*  x1     = (float*)alloc((size_t)BT * D * 4);
  __bf16* x2b    = (__bf16*)alloc((size_t)BT * D * 2);
  __bf16* wqkvb  = (__bf16*)alloc((size_t)3 * D * D * 2);
  __bf16* wprojb = (__bf16*)alloc((size_t)D * D * 2);
  __bf16* w1b    = (__bf16*)alloc((size_t)E * FF * D * 2);
  __bf16* w2b    = (__bf16*)alloc((size_t)E * D * FF * 2);
  float*  probs  = (float*)alloc((size_t)BT * E * 4);
  __bf16* hid    = (__bf16*)alloc((size_t)BT * FF * 2);

  // weight conversions (deterministic, every call)
  k_cvt<<<2048, 256, 0, stream>>>(w_qkv, wqkvb, (size_t)3 * D * D);
  k_cvt<<<1024, 256, 0, stream>>>(w_proj, wprojb, (size_t)D * D);
  k_cvt<<<4096, 256, 0, stream>>>(e_w1, w1b, (size_t)E * FF * D);
  k_cvt<<<4096, 256, 0, stream>>>(e_w2, w2b, (size_t)E * D * FF);

  // ln1 + c
  k_ln<<<BT, 256, 0, stream>>>(x, c, ln1_w, hb, nullptr);
  // qkv
  k_gemm_qkv<<<dim3(3 * D / 64, BT / 128), 256, 0, stream>>>(hb, wqkvb, qb, kb, vb);
  // attention
  k_attn<<<dim3(T / 16, B * H), 32, 0, stream>>>(qb, kb, vb, ob);
  // proj + residual
  k_proj<<<dim3(D / 64, BT / 128), 256, 0, stream>>>(ob, wprojb, x, x1);
  // ln2 -> bf16 activations + f32 output base (out = ln2(x1) + moe)
  k_ln<<<BT, 256, 0, stream>>>(x1, nullptr, ln2_w, x2b, out);
  // router (reads f32 ln2 output from d_out)
  k_router<<<BT, 256, 0, stream>>>(out, router_w, probs);
  // experts
  for (int e = 0; e < E; ++e) {
    k_fc1<<<dim3(FF / 64, BT / 128), 256, 0, stream>>>(
        x2b, w1b + (size_t)e * FF * D, hid);
    k_fc2<<<dim3(D / 64, BT / 128), 256, 0, stream>>>(
        hid, w2b + (size_t)e * D * FF, probs, e, out);
  }
}
